// LSTM_10007273800187
// MI455X (gfx1250) — matrix operand order repaired and checked by reading, not compiled, not run on hardware
//
#include <hip/hip_runtime.h>

#define Bsz  256
#define Tn   512
#define Dn   64
#define Hn   512
#define G4   2048          // 4*H
#define KC   576           // D + H combined reduction dim
#define KT   18            // KC / 32
#define NTIL 128           // G4 / 16
#define BT   16            // batch tile (WMMA M)
#define NBLK (Bsz / BT)    // 16 workgroups
#define NTH  512           // 16 waves of 32
#define ASTR 578           // padded LDS row (odd dword count -> conflict free)

typedef __attribute__((ext_vector_type(16))) __bf16 v16bf;
typedef __attribute__((ext_vector_type(8)))  float  v8f;

__device__ __forceinline__ unsigned short f2bf(float f) {
  unsigned u = __float_as_uint(f);
  u += 0x7FFFu + ((u >> 16) & 1u);       // round-to-nearest-even
  return (unsigned short)(u >> 16);
}

__device__ __forceinline__ float fast_sigmoid(float x) {
  return 1.0f / (1.0f + __expf(-x));
}
__device__ __forceinline__ float fast_tanh(float x) {
  // tanh(x) = 1 - 2/(e^{2x}+1); exact limits at +/-inf via v_exp_f32
  return 1.0f - 2.0f / (__expf(2.0f * x) + 1.0f);
}

// ---------------------------------------------------------------------------
// Kernel A: sample weights (mu + softplus(rho)*eps), convert to bf16, and
// write the combined [w_ih ; w_hh] (576 x 2048) matrix in a pre-swizzled
// per-WMMA-fragment layout: block(kt,nt) = 32 lanes x 8 dwords, lane = N%16 +
// 16*(Kin/16), dword v holds K = 16*(Kin/16) + 2v (+1 in high half).
// Also samples the fp32 bias.
// ---------------------------------------------------------------------------
__global__ void sample_pack(const float* __restrict__ w_ih_mu, const float* __restrict__ w_ih_rho,
                            const float* __restrict__ eps_ih,
                            const float* __restrict__ w_hh_mu, const float* __restrict__ w_hh_rho,
                            const float* __restrict__ eps_hh,
                            const float* __restrict__ b_mu, const float* __restrict__ b_rho,
                            const float* __restrict__ eps_b,
                            unsigned short* __restrict__ wswz, float* __restrict__ bias)
{
  int idx = blockIdx.x * blockDim.x + threadIdx.x;
  const int NW = KC * G4;
  if (idx < NW) {
    int k = idx / G4;
    int n = idx - k * G4;
    float mu, rho, eps;
    if (k < Dn) { int s = k * G4 + n;        mu = w_ih_mu[s]; rho = w_ih_rho[s]; eps = eps_ih[s]; }
    else        { int s = (k - Dn) * G4 + n; mu = w_hh_mu[s]; rho = w_hh_rho[s]; eps = eps_hh[s]; }
    float w = mu + log1pf(expf(rho)) * eps;
    int kt = k >> 5, kin = k & 31, nt = n >> 4, nin = n & 15;
    int lane = nin + ((kin >> 4) << 4);
    int v    = (kin & 15) >> 1;
    int hsel = kin & 1;
    unsigned us = ((((unsigned)(kt * NTIL + nt) * 32u + (unsigned)lane) * 8u) + (unsigned)v) * 2u + (unsigned)hsel;
    wswz[us] = f2bf(w);
  } else if (idx < NW + G4) {
    int n = idx - NW;
    bias[n] = b_mu[n] + log1pf(expf(b_rho[n])) * eps_b[n];
  }
}

// ---------------------------------------------------------------------------
// Kernel B: persistent LSTM. One workgroup owns 16 batch rows for all 512
// timesteps. Per step: stage x_t (bf16) into LDS next to h (A = [x|h],
// 16x576), single-pass GEMM of all 8 output tiles per wave via
// v_wmma_f32_16x16x32_bf16 with explicit one-kt double buffering of A/B
// fragments, then fp32 cell update from LDS-staged gates in two halves.
// ---------------------------------------------------------------------------
__global__ void __launch_bounds__(NTH, 1)
lstm_persist(const float* __restrict__ x, const unsigned short* __restrict__ wswz,
             const float* __restrict__ bias, const float* __restrict__ lin_w,
             const float* __restrict__ lin_b, float* __restrict__ out)
{
  __shared__ unsigned short sA[BT][ASTR];   // bf16 [x_t | h], padded rows
  __shared__ float          sG[BT][1024];   // gate pre-activations, one half
  __shared__ float          sC[BT][Hn];     // cell state fp32 (h fp32 at the end)

  const int tid  = threadIdx.x;
  const int lane = tid & 31;
  const int wv   = tid >> 5;            // wave 0..15
  const int b0   = blockIdx.x * BT;
  const int r16  = lane & 15;           // WMMA row / column-in-tile
  const int kb   = (lane >> 4) << 4;    // 0 or 16 (A-fragment K base)

  // init h = 0 (bf16 in sA), c = 0
  for (int e = tid; e < BT * Hn; e += NTH) {
    int r = e >> 9, j = e & (Hn - 1);
    sA[r][Dn + j] = 0;
    sC[r][j] = 0.0f;
  }
  __syncthreads();

  union U { unsigned u[8]; v16bf v; };

  const unsigned short* arow = &sA[r16][0];
  const uint4* bbase = (const uint4*)wswz + ((unsigned)(wv * 32 + lane) * 2u);

  // fragment loaders (kt-indexed)
  auto loadA = [&](U& dst, int kt) {
    const int k0 = kt * 32 + kb;
    #pragma unroll
    for (int vv = 0; vv < 4; ++vv)
      dst.u[vv] = *(const unsigned*)(arow + k0 + 2 * vv);
    #pragma unroll
    for (int vv = 0; vv < 4; ++vv)
      dst.u[4 + vv] = *(const unsigned*)(arow + k0 + 8 + 2 * vv);
  };
  auto loadB = [&](U (&dst)[2][4], int kt) {
    const uint4* bp = bbase + (unsigned)kt * (NTIL * 64u);   // 128 tiles * 32 lanes * 2 uint4
    #pragma unroll
    for (int h = 0; h < 2; ++h)
      #pragma unroll
      for (int q = 0; q < 4; ++q) {
        const uint4* pp = bp + (q * 2048 + h * 1024);        // nt = q*32 + h*16 + wv
        uint4 lo = pp[0], hi = pp[1];
        dst[h][q].u[0] = lo.x; dst[h][q].u[1] = lo.y;
        dst[h][q].u[2] = lo.z; dst[h][q].u[3] = lo.w;
        dst[h][q].u[4] = hi.x; dst[h][q].u[5] = hi.y;
        dst[h][q].u[6] = hi.z; dst[h][q].u[7] = hi.w;
      }
  };

  for (int t = 0; t < Tn; ++t) {
    // ---- refresh x_t into sA cols [0,64) ----
    for (int e = tid; e < BT * Dn; e += NTH) {
      int r = e >> 6, d = e & (Dn - 1);
      sA[r][d] = f2bf(x[(size_t)(b0 + r) * (Tn * Dn) + (size_t)t * Dn + d]);
    }
    __syncthreads();

    // ---- single-pass GEMM, one-kt-deep software pipeline ----
    v8f acc[2][4] = {};
    U af[2];
    U bfr[2][2][4];
    loadA(af[0], 0);
    loadB(bfr[0], 0);
    #pragma unroll 2
    for (int kt = 0; kt < KT; ++kt) {
      const int cur = kt & 1, nxt = cur ^ 1;
      const int ktn = (kt + 1 < KT) ? kt + 1 : kt;   // last iter: harmless reload
      loadA(af[nxt], ktn);                            // prefetch next kt
      loadB(bfr[nxt], ktn);
      #pragma unroll
      for (int h = 0; h < 2; ++h)
        #pragma unroll
        for (int q = 0; q < 4; ++q)
          acc[h][q] = __builtin_amdgcn_wmma_f32_16x16x32_bf16(
              false, af[cur].v, false, bfr[cur][h][q].v, (short)0, acc[h][q], false, false);
    }

    // ---- stage + cell-update in two 1024-column halves ----
    #pragma unroll
    for (int h = 0; h < 2; ++h) {
      const int ht = h * 16 + wv;
      #pragma unroll
      for (int q = 0; q < 4; ++q) {
        const int ncol = (q * 32 + ht) * 16 + r16;       // global gate column
        const float bs = bias[ncol];
        const int lcol = q * 256 + wv * 16 + r16;        // half-local column
        const int m0   = (lane >> 4) * 8;
        #pragma unroll
        for (int rr = 0; rr < 8; ++rr)
          sG[m0 + rr][lcol] = acc[h][q][rr] + bs;
      }
      __syncthreads();
      {
        const int jl = tid & 255;
        const int rb = (tid >> 8) * 8;
        const int j  = h * 256 + jl;
        for (int rr = 0; rr < 8; ++rr) {
          const int b = rb + rr;
          float gi = sG[b][0 * 256 + jl];
          float gf = sG[b][1 * 256 + jl];
          float gg = sG[b][2 * 256 + jl];
          float go = sG[b][3 * 256 + jl];
          float ig = fast_sigmoid(gi);
          float fg = fast_sigmoid(gf);
          float gv = fast_tanh(gg);
          float og = fast_sigmoid(go);
          float cn = fg * sC[b][j] + ig * gv;
          float hn = og * fast_tanh(cn);
          sA[b][Dn + j] = f2bf(hn);                 // GEMM for this t is done
          sC[b][j] = (t == Tn - 1) ? hn : cn;       // last step: keep h fp32
        }
      }
      __syncthreads();
    }
  }

  // ---- final linear: out[b] = sum_j h[b][j] * lin_w[j] + lin_b ----
  {
    const int j = tid;          // NTH == Hn == 512
    const float lw = lin_w[j];
    for (int b = 0; b < BT; ++b) sG[b][j] = sC[b][j] * lw;
  }
  __syncthreads();
  if (tid < BT) {
    float s = 0.0f;
    for (int j = 0; j < Hn; ++j) s += sG[tid][j];
    out[b0 + tid] = s + lin_b[0];
  }
}

extern "C" void kernel_launch(void* const* d_in, const int* in_sizes, int n_in,
                              void* d_out, int out_size, void* d_ws, size_t ws_size,
                              hipStream_t stream) {
  (void)in_sizes; (void)n_in; (void)out_size; (void)ws_size;
  const float* x        = (const float*)d_in[0];
  const float* w_ih_mu  = (const float*)d_in[1];
  const float* w_ih_rho = (const float*)d_in[2];
  const float* eps_ih   = (const float*)d_in[3];
  const float* w_hh_mu  = (const float*)d_in[4];
  const float* w_hh_rho = (const float*)d_in[5];
  const float* eps_hh   = (const float*)d_in[6];
  const float* b_mu     = (const float*)d_in[7];
  const float* b_rho    = (const float*)d_in[8];
  const float* eps_b    = (const float*)d_in[9];
  const float* lin_w    = (const float*)d_in[10];
  const float* lin_b    = (const float*)d_in[11];

  unsigned short* wswz = (unsigned short*)d_ws;                       // 576*2048 bf16 = 2.25 MB
  float* bias = (float*)((char*)d_ws + (size_t)KC * G4 * 2);          // 2048 fp32

  const int total = KC * G4 + G4;
  sample_pack<<<(total + 255) / 256, 256, 0, stream>>>(
      w_ih_mu, w_ih_rho, eps_ih, w_hh_mu, w_hh_rho, eps_hh,
      b_mu, b_rho, eps_b, wswz, bias);

  lstm_persist<<<NBLK, NTH, 0, stream>>>(x, wswz, bias, lin_w, lin_b, (float*)d_out);
}
